// SimplePose_63067299775013
// MI455X (gfx1250) — matrix operand order repaired and checked by reading, not compiled, not run on hardware
//
#include <hip/hip_runtime.h>
#include <hip/hip_bf16.h>
#include <stdint.h>

typedef __bf16 bf16;
typedef __attribute__((ext_vector_type(16))) __bf16 v16bf;
typedef __attribute__((ext_vector_type(8)))  __bf16 v8bf;
typedef __attribute__((ext_vector_type(8)))  float  v8f;

#define KPT 17
#define COUT 256

__device__ __forceinline__ v16bf cat16(v8bf lo, v8bf hi) {
    return __builtin_shufflevector(lo, hi, 0,1,2,3,4,5,6,7,8,9,10,11,12,13,14,15);
}
__device__ __forceinline__ v8f wmma_bf16(v16bf a, v16bf b, v8f c) {
    return __builtin_amdgcn_wmma_f32_16x16x32_bf16(false, a, false, b, (short)0, c, false, false);
}

// ---------------- fp32 -> bf16 elementwise ----------------
__global__ void f32_to_bf16_kernel(const float* __restrict__ in, bf16* __restrict__ out, int n) {
    int i = blockIdx.x * blockDim.x + threadIdx.x;
    if (i < n) out[i] = (bf16)in[i];
}

// ---------------- BN fold: s = g*rsqrt(v+eps), t = b - m*s ----------------
__global__ void bn_fold_kernel(const float* __restrict__ g, const float* __restrict__ b,
                               const float* __restrict__ m, const float* __restrict__ v,
                               float* __restrict__ s, float* __restrict__ t, int C) {
    int i = blockIdx.x * blockDim.x + threadIdx.x;
    if (i < C) {
        float sc = g[i] * rsqrtf(v[i] + 1e-5f);
        s[i] = sc;
        t[i] = b[i] - m[i] * sc;
    }
}

// ---------------- weight pack: w[4][4][Cin][256] f32 -> WMMA B-fragment tiles bf16 ----
// packed index: (((tap*(Cin/32) + cic)*16 + coTile)*32 + lane)*16 + e
// B fragment layout (32x16, bf16): lane<16: N=lane,    K=e
//                                  lane>=16: N=lane-16, K=16+e
__global__ void pack_w_kernel(const float* __restrict__ w, bf16* __restrict__ p, int Cin) {
    long i = (long)blockIdx.x * blockDim.x + threadIdx.x;
    long total = 16L * Cin * COUT;
    if (i >= total) return;
    int e      = (int)(i & 15);
    int lane   = (int)((i >> 4) & 31);
    int coTile = (int)((i >> 9) & 15);
    long r     = i >> 13;
    int nChunk = Cin >> 5;
    int cic = (int)(r % nChunk);
    int tap = (int)(r / nChunk);             // tap = kh*4+kw
    int n  = lane & 15;
    int k  = ((lane >> 4) << 4) | e;
    int ci = (cic << 5) + k;
    int co = (coTile << 4) + n;
    p[i] = (bf16)w[((long)tap * Cin + ci) * COUT + co];
}

// ---------------- deconv (ConvTranspose k4 s2 p1) + BN + ReLU via WMMA ----------------
// Each wave computes a 32-pixel x 64-channel tile of one parity quadrant.
// No LDS, no barriers: A fragments load straight from global (L2-resident bf16
// activations) as two b128 per lane per fragment; B fragments are pre-packed.
// grid.x = (B*HIN*WIN/32)*4 / 8 waves, grid.y = 4 quadrants, block = 256.
template<int HIN, int WIN, int CIN>
__global__ __launch_bounds__(256)
void deconv_wmma_kernel(const bf16* __restrict__ xin, const bf16* __restrict__ wpk,
                        const float* __restrict__ sArr, const float* __restrict__ tArr,
                        bf16* __restrict__ yout) {
    constexpr int QHW = HIN * WIN;
    constexpr int OH = 2 * HIN, OW = 2 * WIN;
    constexpr int NCH = CIN / 32;

    const int po = blockIdx.y >> 1, pw = blockIdx.y & 1;
    const int lane = threadIdx.x & 31;
    const int wv   = threadIdx.x >> 5;
    const int wg   = blockIdx.x * 8 + wv;
    const int coT  = wg & 3;                  // 64-channel group
    const int base = (wg >> 2) * 32;          // 32 quadrant pixels

    const int m = lane & 15, half = lane >> 4;

    // decompose this lane's two A rows (pixels base+m, base+16+m)
    const int r0 = base + m,        r1 = base + 16 + m;
    const int b0 = r0 / QHW, q0 = r0 % QHW, qh0 = q0 / WIN, qw0 = q0 % WIN;
    const int b1 = r1 / QHW, q1 = r1 % QHW, qh1 = q1 / WIN, qw1 = q1 % WIN;

    v8f acc[2][4] = {};

    #pragma unroll
    for (int tap = 0; tap < 4; ++tap) {
        const int dh = tap >> 1, dw = tap & 1;
        const int ih0 = qh0 + dh + po - 1, iw0 = qw0 + dw + pw - 1;
        const int ih1 = qh1 + dh + po - 1, iw1 = qw1 + dw + pw - 1;
        const bool ok0 = ((unsigned)ih0 < (unsigned)HIN) && ((unsigned)iw0 < (unsigned)WIN);
        const bool ok1 = ((unsigned)ih1 < (unsigned)HIN) && ((unsigned)iw1 < (unsigned)WIN);
        const bf16* p0 = xin + ((long)(b0 * HIN + ih0) * WIN + iw0) * CIN + half * 16;
        const bf16* p1 = xin + ((long)(b1 * HIN + ih1) * WIN + iw1) * CIN + half * 16;
        const int tapLin = (po + 2 * dh) * 4 + (pw + 2 * dw);
        const bf16* bb = wpk + (long)tapLin * NCH * 8192 + coT * 4 * 512 + lane * 16;

        for (int cic = 0; cic < NCH; ++cic) {
            // A fragments (two contiguous 16B chunks/lane: k = 16*half + e, same order as packed B)
            v8bf a0l = {}, a0h = {}, a1l = {}, a1h = {};
            if (ok0) { a0l = *(const v8bf*)(p0 + cic * 32);
                       a0h = *(const v8bf*)(p0 + cic * 32 + 8); }
            if (ok1) { a1l = *(const v8bf*)(p1 + cic * 32);
                       a1h = *(const v8bf*)(p1 + cic * 32 + 8); }
            v16bf a0 = cat16(a0l, a0h);
            v16bf a1 = cat16(a1l, a1h);

            const bf16* bp = bb + (long)cic * 8192;
            __builtin_prefetch(bp + 8192, 0, 1);   // next K-step's B tiles -> global_prefetch
            #pragma unroll
            for (int j = 0; j < 4; ++j) {
                v8bf bl = *(const v8bf*)(bp + j * 512);
                v8bf bh = *(const v8bf*)(bp + j * 512 + 8);
                v16bf bfrag = cat16(bl, bh);
                acc[0][j] = wmma_bf16(a0, bfrag, acc[0][j]);
                acc[1][j] = wmma_bf16(a1, bfrag, acc[1][j]);
            }
        }
    }

    // Epilogue: folded BN + ReLU, store bf16 NHWC
    #pragma unroll
    for (int i = 0; i < 2; ++i) {
        #pragma unroll
        for (int j = 0; j < 4; ++j) {
            const int co = coT * 64 + j * 16 + m;
            const float sc = sArr[co], bi = tArr[co];
            #pragma unroll
            for (int v = 0; v < 8; ++v) {
                const int row = base + i * 16 + half * 8 + v;   // C/D: lane<16 M=v, lane>=16 M=v+8
                const int b2 = row / QHW, rr = row % QHW;
                const int oh = 2 * (rr / WIN) + po, ow = 2 * (rr % WIN) + pw;
                float val = acc[i][j][v] * sc + bi;
                val = val > 0.f ? val : 0.f;
                yout[((long)(b2 * OH + oh) * OW + ow) * COUT + co] = (bf16)val;
            }
        }
    }
}

// ---------------- final 1x1 conv: heatmap[p][k] = sum_c y[p][c]*wf[c][k] + bf[k] ----------
__global__ __launch_bounds__(256)
void conv1x1_kernel(const bf16* __restrict__ y, const float* __restrict__ wf,
                    const float* __restrict__ bfv, float* __restrict__ hm, int nPix) {
    __shared__ float wfs[COUT * KPT];
    __shared__ float bfs[KPT];
    for (int i = threadIdx.x; i < COUT * KPT; i += 256) wfs[i] = wf[i];
    if (threadIdx.x < KPT) bfs[threadIdx.x] = bfv[threadIdx.x];
    __syncthreads();
    int p = blockIdx.x * 256 + threadIdx.x;
    if (p >= nPix) return;
    float acc[KPT];
    for (int k = 0; k < KPT; ++k) acc[k] = bfs[k];
    const bf16* yp = y + (long)p * COUT;
    for (int c = 0; c < COUT; ++c) {
        float yv = (float)yp[c];
        const float* wr = wfs + c * KPT;
        #pragma unroll
        for (int k = 0; k < KPT; ++k) acc[k] = fmaf(yv, wr[k], acc[k]);
    }
    float* o = hm + (long)p * KPT;
    for (int k = 0; k < KPT; ++k) o[k] = acc[k];
}

// ---------------- heatmap max detection (argmax w/ first-index ties + subpixel) ---------
__global__ __launch_bounds__(256)
void hm_det_kernel(const float* __restrict__ hm, float* __restrict__ out, int H, int W) {
    const int HW = H * W;
    const int b = blockIdx.x / KPT;
    const int k = blockIdx.x % KPT;
    __shared__ float sv[256];
    __shared__ int   si[256];
    float bv = -__builtin_inff();
    int   bi = 0;
    for (int i = threadIdx.x; i < HW; i += 256) {
        float v = hm[((long)b * HW + i) * KPT + k];
        if (v > bv) { bv = v; bi = i; }
    }
    sv[threadIdx.x] = bv; si[threadIdx.x] = bi;
    __syncthreads();
    for (int s = 128; s > 0; s >>= 1) {
        if (threadIdx.x < s) {
            float ov = sv[threadIdx.x + s]; int oi = si[threadIdx.x + s];
            if (ov > sv[threadIdx.x] || (ov == sv[threadIdx.x] && oi < si[threadIdx.x])) {
                sv[threadIdx.x] = ov; si[threadIdx.x] = oi;
            }
        }
        __syncthreads();
    }
    if (threadIdx.x == 0) {
        float score = sv[0]; int idx = si[0];
        int px = 0, py = 0;
        if (score > 0.f) { px = idx % W; py = idx / W; }
        bool inner = (px > 0) && (px < W - 1) && (py > 0) && (py < H - 1);
        float dx = 0.f, dy = 0.f;
        if (inner) {
            int basei = py * W + px;
            auto g = [&](int off) {
                int i2 = basei + off;
                i2 = i2 < 0 ? 0 : (i2 > HW - 1 ? HW - 1 : i2);
                return hm[((long)b * HW + i2) * KPT + k];
            };
            float d1 = g(1) - g(-1);
            float d2 = g(W) - g(-W);
            dx = d1 > 0.f ? 0.25f : (d1 < 0.f ? -0.25f : 0.f);
            dy = d2 > 0.f ? 0.25f : (d2 < 0.f ? -0.25f : 0.f);
        }
        float* o = out + ((long)b * KPT + k) * 3;
        o[0] = (float)px + dx;
        o[1] = (float)py + dy;
        o[2] = score;
    }
}

// ---------------- launcher ----------------
static inline size_t alignUp256(size_t x) { return (x + 255) & ~(size_t)255; }

extern "C" void kernel_launch(void* const* d_in, const int* in_sizes, int n_in,
                              void* d_out, int out_size, void* d_ws, size_t ws_size,
                              hipStream_t stream) {
    (void)in_sizes; (void)n_in; (void)out_size; (void)ws_size;
    const int B = 64;
    const int H0 = 8, W0 = 6, C0 = 2048;

    const float* x  = (const float*)d_in[0];
    const float* w1 = (const float*)d_in[1];
    const float* g1 = (const float*)d_in[2];
    const float* b1 = (const float*)d_in[3];
    const float* m1 = (const float*)d_in[4];
    const float* v1 = (const float*)d_in[5];
    const float* w2 = (const float*)d_in[6];
    const float* g2 = (const float*)d_in[7];
    const float* b2 = (const float*)d_in[8];
    const float* m2 = (const float*)d_in[9];
    const float* v2 = (const float*)d_in[10];
    const float* w3 = (const float*)d_in[11];
    const float* g3 = (const float*)d_in[12];
    const float* b3 = (const float*)d_in[13];
    const float* m3 = (const float*)d_in[14];
    const float* v3 = (const float*)d_in[15];
    const float* wf = (const float*)d_in[16];
    const float* bf = (const float*)d_in[17];
    float* outp = (float*)d_out;

    char* basep = (char*)d_ws;
    size_t off = 0;
    auto alloc = [&](size_t bytes) { char* p = basep + off; off = alignUp256(off + bytes); return p; };

    const long nX = (long)B * H0 * W0 * C0;
    bf16*  xb  = (bf16*)alloc(nX * 2);
    bf16*  w1p = (bf16*)alloc(16L * C0  * COUT * 2);
    bf16*  w2p = (bf16*)alloc(16L * 256 * COUT * 2);
    bf16*  w3p = (bf16*)alloc(16L * 256 * COUT * 2);
    float* s1  = (float*)alloc(COUT * 4);  float* t1 = (float*)alloc(COUT * 4);
    float* s2  = (float*)alloc(COUT * 4);  float* t2 = (float*)alloc(COUT * 4);
    float* s3  = (float*)alloc(COUT * 4);  float* t3 = (float*)alloc(COUT * 4);
    bf16*  y1  = (bf16*)alloc((long)B * 16 * 12 * COUT * 2);
    bf16*  y2  = (bf16*)alloc((long)B * 32 * 24 * COUT * 2);
    bf16*  y3  = (bf16*)alloc((long)B * 64 * 48 * COUT * 2);
    float* hm  = (float*)alloc((long)B * 64 * 48 * KPT * 4);

    // 1) convert x to bf16
    f32_to_bf16_kernel<<<(unsigned)((nX + 255) / 256), 256, 0, stream>>>(x, xb, (int)nX);

    // 2) pack weights into WMMA B-fragment layout
    {
        long t = 16L * C0 * COUT;
        pack_w_kernel<<<(unsigned)((t + 255) / 256), 256, 0, stream>>>(w1, w1p, C0);
        t = 16L * 256 * COUT;
        pack_w_kernel<<<(unsigned)((t + 255) / 256), 256, 0, stream>>>(w2, w2p, 256);
        pack_w_kernel<<<(unsigned)((t + 255) / 256), 256, 0, stream>>>(w3, w3p, 256);
    }

    // 3) BN folds
    bn_fold_kernel<<<1, 256, 0, stream>>>(g1, b1, m1, v1, s1, t1, COUT);
    bn_fold_kernel<<<1, 256, 0, stream>>>(g2, b2, m2, v2, s2, t2, COUT);
    bn_fold_kernel<<<1, 256, 0, stream>>>(g3, b3, m3, v3, s3, t3, COUT);

    // 4) deconv stack: grid.x = (#quad pixels / 32) * 4 co-groups / 8 waves per block
    deconv_wmma_kernel<8, 6, 2048><<<dim3((B * 8 * 6 / 32) * 4 / 8, 4), 256, 0, stream>>>(
        xb, w1p, s1, t1, y1);                                   // -> [64,16,12,256]
    deconv_wmma_kernel<16, 12, 256><<<dim3((B * 16 * 12 / 32) * 4 / 8, 4), 256, 0, stream>>>(
        y1, w2p, s2, t2, y2);                                   // -> [64,32,24,256]
    deconv_wmma_kernel<32, 24, 256><<<dim3((B * 32 * 24 / 32) * 4 / 8, 4), 256, 0, stream>>>(
        y2, w3p, s3, t3, y3);                                   // -> [64,64,48,256]

    // 5) final 1x1 conv to 17-channel heatmap
    const int nPix = B * 64 * 48;
    conv1x1_kernel<<<(nPix + 255) / 256, 256, 0, stream>>>(y3, wf, bf, hm, nPix);

    // 6) heatmap max detection -> [B, 17, 3]
    hm_det_kernel<<<B * KPT, 256, 0, stream>>>(hm, outp, 64, 48);
}
